// Decoder_81183471829595
// MI455X (gfx1250) — matrix operand order repaired and checked by reading, not compiled, not run on hardware
//
#include <hip/hip_runtime.h>
#include <hip/hip_bf16.h>

// ---------------------------------------------------------------------------
// Tacotron decoder for MI455X (gfx1250, wave32).
//   * Prenet + keys hoisted out of the scan as full 16-row WMMA bf16 GEMMs.
//   * Weights pre-packed into WMMA B-fragment tile layout (bf16): 32B/lane
//     contiguous loads per 16x16x32 k-tile.
//   * Scan: 64 persistent WGs (one independent batch recurrence each),
//     512 threads / 16 waves. All dense layers via v_wmma_f32_16x16x32_bf16
//     with M=1 (row-0 lanes carry data, others zero-selected; broadcast LDS
//     reads). Independent GRU input/recurrent GEMVs run concurrently on
//     split wave groups. Keys slab (256KB bf16) lives in LDS for all 400
//     steps; softmax uses wave32 shuffle reductions. All GEMV shapes are
//     compile-time so k-tile loops fully unroll (loads pipelined ahead of
//     the serial wmma accumulation chain).
// ---------------------------------------------------------------------------

typedef __bf16 v16bf __attribute__((ext_vector_type(16)));
typedef float  v8f   __attribute__((ext_vector_type(8)));

#define NB    64
#define TDEC  400
#define TENC  512
#define DIM   256
#define PREN  128
#define NOUT  400

// ---------------------------------------------------------------------------
// Full 16-row GEMM tile engine (prep kernels). A: LDS fp32 [16 x lda].
// Wp: packed bf16 tiles, tile (nt,kt) at (nt*KT+kt)*512, lane data at lane*16.
// ---------------------------------------------------------------------------
template <int K, int N>
__device__ __forceinline__ void gemm_tiles16(const float* __restrict__ A, int lda,
                                             const __bf16* __restrict__ Wp,
                                             const float* __restrict__ bias,
                                             float* __restrict__ C, int ldc) {
  constexpr int KT = K / 32;
  constexpr int NT = N / 16;
  const int tid    = threadIdx.x;
  const int lane   = tid & 31;
  const int wave   = tid >> 5;
  const int nwaves = blockDim.x >> 5;
  const int row    = lane & 15;
  const int hi     = lane >> 4;
  for (int nt = wave; nt < NT; nt += nwaves) {
    v8f acc;
    const float bv = bias ? bias[nt * 16 + row] : 0.0f;
#pragma unroll
    for (int r = 0; r < 8; ++r) acc[r] = bv;
    const __bf16* wt   = Wp + (size_t)(nt * KT) * 512 + lane * 16;
    const float*  arow = A + row * lda + hi * 8;
#pragma unroll
    for (int kt = 0; kt < KT; ++kt) {
      v16bf a;
      const float* ar = arow + kt * 32;
#pragma unroll
      for (int i = 0; i < 8; ++i) {
        a[i]     = (__bf16)ar[i];       // K = kt*32 + hi*8 + i
        a[i + 8] = (__bf16)ar[16 + i];  // K = kt*32 + 16 + hi*8 + i
      }
      v16bf b = *(const v16bf*)(wt + (size_t)kt * 512);
      acc = __builtin_amdgcn_wmma_f32_16x16x32_bf16(false, a, false, b,
                                                    (short)0, acc, false, false);
    }
#pragma unroll
    for (int r = 0; r < 8; ++r)
      C[(r + hi * 8) * ldc + nt * 16 + row] = acc[r];
  }
}

// ---------------------------------------------------------------------------
// M=1 GEMV via WMMA: vec (LDS fp32, length K) x Wp -> outv[N] (row 0 only).
// Waves [W0, W0+NW) participate; lanes with (lane&15)!=0 contribute zero A
// rows (broadcast LDS reads + select; no staging buffer). Fully unrolled.
// ---------------------------------------------------------------------------
template <int K, int N, int W0, int NW>
__device__ __forceinline__ void gemv_row0(const float* __restrict__ vec,
                                          const __bf16* __restrict__ Wp,
                                          const float* __restrict__ bias,
                                          float* __restrict__ outv) {
  constexpr int KT = K / 32;
  constexpr int NT = N / 16;
  const int lane = threadIdx.x & 31;
  const int wave = threadIdx.x >> 5;
  if (wave < W0 || wave >= W0 + NW) return;
  const int  col = lane & 15;
  const int  hi  = lane >> 4;
  const bool r0  = (col == 0);   // lanes 0 and 16 carry A row 0
  for (int nt = wave - W0; nt < NT; nt += NW) {
    v8f acc;
    const float bv = bias ? bias[nt * 16 + col] : 0.0f;
#pragma unroll
    for (int r = 0; r < 8; ++r) acc[r] = bv;
    const __bf16* wt = Wp + (size_t)(nt * KT) * 512 + lane * 16;
#pragma unroll
    for (int kt = 0; kt < KT; ++kt) {
      const float* ar = vec + kt * 32 + hi * 8;  // same addr across lanes -> broadcast
      v16bf a;
#pragma unroll
      for (int i = 0; i < 8; ++i) {
        a[i]     = r0 ? (__bf16)ar[i]      : (__bf16)0.0f;
        a[i + 8] = r0 ? (__bf16)ar[16 + i] : (__bf16)0.0f;
      }
      v16bf b = *(const v16bf*)(wt + (size_t)kt * 512);
      acc = __builtin_amdgcn_wmma_f32_16x16x32_bf16(false, a, false, b,
                                                    (short)0, acc, false, false);
    }
    if (lane < 16) outv[nt * 16 + lane] = acc[0];  // M=0 row
  }
}

// ---------------------------------------------------------------------------
// Pack fp32 [K x N] weight into WMMA B-fragment tile layout (bf16), zero-pad
// K up to KT*32. Element i of lane L in tile (kt,nt):
//   k = kt*32 + (i>>3)*16 + (L>>4)*8 + (i&7),  n = nt*16 + (L&15)
// ---------------------------------------------------------------------------
__global__ void pack_kernel(const float* __restrict__ src, __bf16* __restrict__ dst,
                            int K, int KT, int N, int total) {
  int idx = blockIdx.x * blockDim.x + threadIdx.x;
  if (idx >= total) return;
  int e    = idx & 15;
  int lane = (idx >> 4) & 31;
  int tile = idx >> 9;
  int kt   = tile % KT;
  int nt   = tile / KT;
  int k = kt * 32 + ((e >> 3) << 4) + ((lane >> 4) << 3) + (e & 7);
  int n = nt * 16 + (lane & 15);
  float v = (k < K) ? src[(size_t)k * N + n] : 0.0f;
  dst[idx] = (__bf16)v;
}

// Prenet layer 1: P1 = relu(dec @ W1 + b1); K=400 padded to 416.
__global__ void __launch_bounds__(512) prenet1_kernel(
    const float* __restrict__ dec, const __bf16* __restrict__ W1p,
    const float* __restrict__ b1, __bf16* __restrict__ P1) {
  __shared__ float A[16 * 416];
  __shared__ float C[16 * 256];
  const int g0 = blockIdx.x * 16;
  for (int idx = threadIdx.x; idx < 16 * 416; idx += 512) {
    int r = idx / 416, k = idx - r * 416;
    A[idx] = (k < NOUT) ? dec[(size_t)(g0 + r) * NOUT + k] : 0.0f;
  }
  __syncthreads();
  gemm_tiles16<416, 256>(A, 416, W1p, b1, C, 256);
  __syncthreads();
  for (int idx = threadIdx.x; idx < 16 * 256; idx += 512) {
    int r = idx >> 8, n = idx & 255;
    float v = C[idx];
    P1[(size_t)(g0 + r) * 256 + n] = (__bf16)(v > 0.0f ? v : 0.0f);
  }
}

// Prenet layer 2: P = relu(P1 @ W2 + b2)
__global__ void __launch_bounds__(512) prenet2_kernel(
    const __bf16* __restrict__ P1, const __bf16* __restrict__ W2p,
    const float* __restrict__ b2, __bf16* __restrict__ P) {
  __shared__ float A[16 * 256];
  __shared__ float C[16 * 128];
  const int g0 = blockIdx.x * 16;
  for (int idx = threadIdx.x; idx < 16 * 256; idx += 512) {
    int r = idx >> 8, k = idx & 255;
    A[idx] = (float)P1[(size_t)(g0 + r) * 256 + k];
  }
  __syncthreads();
  gemm_tiles16<256, 128>(A, 256, W2p, b2, C, 128);
  __syncthreads();
  for (int idx = threadIdx.x; idx < 16 * 128; idx += 512) {
    int r = idx >> 7, n = idx & 127;
    float v = C[idx];
    P[(size_t)(g0 + r) * 128 + n] = (__bf16)(v > 0.0f ? v : 0.0f);
  }
}

// keys = memory @ Wm  (bf16; streamed into each WG's LDS by the scan)
__global__ void __launch_bounds__(512) keys_kernel(
    const float* __restrict__ memory, const __bf16* __restrict__ Wmp,
    __bf16* __restrict__ keysb) {
  __shared__ float A[16 * 256];
  __shared__ float C[16 * 256];
  const int g0 = blockIdx.x * 16;
  for (int idx = threadIdx.x; idx < 16 * 256; idx += 512) {
    int r = idx >> 8, k = idx & 255;
    A[idx] = memory[(size_t)(g0 + r) * 256 + k];
  }
  __syncthreads();
  gemm_tiles16<256, 256>(A, 256, Wmp, nullptr, C, 256);
  __syncthreads();
  for (int idx = threadIdx.x; idx < 16 * 256; idx += 512) {
    int r = idx >> 8, n = idx & 255;
    keysb[(size_t)(g0 + r) * 256 + n] = (__bf16)C[idx];
  }
}

__device__ __forceinline__ float sigmoidf_(float x) {
  return 1.0f / (1.0f + __expf(-x));
}

// ---------------------------------------------------------------------------
// Persistent decoder scan: one WG per batch row, 512 threads (16 waves).
// Dynamic LDS: entire keys slab for this batch row (512x256 bf16 = 256KB).
// ---------------------------------------------------------------------------
__global__ void __launch_bounds__(512) decoder_kernel(
    const __bf16* __restrict__ P, const __bf16* __restrict__ keysb,
    const float* __restrict__ memory,
    const __bf16* __restrict__ k0p, const float* __restrict__ bi0,
    const __bf16* __restrict__ r0p, const float* __restrict__ br0,
    const __bf16* __restrict__ k1p, const float* __restrict__ bi1,
    const __bf16* __restrict__ r1p, const float* __restrict__ br1,
    const __bf16* __restrict__ Wqp, const float* __restrict__ vvec,
    const __bf16* __restrict__ Wap, const __bf16* __restrict__ Wop,
    const float* __restrict__ bo, float* __restrict__ out) {
  __shared__ float cin[384];
  __shared__ float gx[768], gh[768];
  __shared__ float q[256], sc[512], red[512];
  __shared__ float h1ctx[512];           // [0..255]=h1, [256..511]=ctx
  __shared__ float h0[256], att[256];
  __shared__ float vbuf[256];
  extern __shared__ __bf16 kdyn[];       // 512 x 256 bf16 keys slab

  const int tid  = threadIdx.x;
  const int lane = tid & 31;
  const int wave = tid >> 5;
  const int b    = blockIdx.x;

  float* h1  = h1ctx;
  float* ctx = h1ctx + 256;

  // one-time: state init, v -> LDS, keys slab -> LDS (coalesced 16B chunks)
  if (tid < 256) {
    h0[tid] = 0.0f; h1[tid] = 0.0f; att[tid] = 0.0f;
    vbuf[tid] = vvec[tid];
  }
  {
    const int4* src = (const int4*)(keysb + (size_t)b * TENC * DIM);
    int4* dst = (int4*)kdyn;
    for (int i = tid; i < TENC * DIM * 2 / 16; i += 512) dst[i] = src[i];
  }
  __syncthreads();

  const float* mb = memory + (size_t)b * TENC * DIM;
  const int rot = (lane * 8) & 255;      // bank-conflict skew for score loop

  for (int t = 0; t < TDEC; ++t) {
    // 1) cin = [prenet(x_t), att]
    const __bf16* xr = P + ((size_t)b * TDEC + t) * PREN;
    if (tid < 384) cin[tid] = (tid < PREN) ? (float)xr[tid] : att[tid - PREN];
    __syncthreads();
    // 2) GRU0 gates: gx = cin@k0+bi0 (waves 0-7) || gh = h0@r0+br0 (waves 8-15)
    gemv_row0<384, 768, 0, 8>(cin, k0p, bi0, gx);
    gemv_row0<256, 768, 8, 8>(h0, r0p, br0, gh);
    __syncthreads();
    // 3) GRU0 combine (Keras reset_after)
    if (tid < 256) {
      float z  = sigmoidf_(gx[tid] + gh[tid]);
      float r  = sigmoidf_(gx[256 + tid] + gh[256 + tid]);
      float hc = tanhf(gx[512 + tid] + r * gh[512 + tid]);
      h0[tid]  = z * h0[tid] + (1.0f - z) * hc;
    }
    __syncthreads();
    // 4) GRU1 gates (concurrent pair)
    gemv_row0<256, 768, 0, 8>(h0, k1p, bi1, gx);
    gemv_row0<256, 768, 8, 8>(h1, r1p, br1, gh);
    __syncthreads();
    // 5) GRU1 combine
    if (tid < 256) {
      float z  = sigmoidf_(gx[tid] + gh[tid]);
      float r  = sigmoidf_(gx[256 + tid] + gh[256 + tid]);
      float hc = tanhf(gx[512 + tid] + r * gh[512 + tid]);
      h1[tid]  = z * h1[tid] + (1.0f - z) * hc;
    }
    __syncthreads();
    // 6) query: q = h1 @ Wq
    gemv_row0<256, 256, 0, 16>(h1, Wqp, nullptr, q);
    __syncthreads();
    // 7) Bahdanau score (one encoder slot per thread) + shuffle softmax
    float s = 0.0f;
    {
      const __bf16* kr = kdyn + (size_t)tid * DIM;
#pragma unroll 4
      for (int j = 0; j < DIM; ++j) {
        int d = (j + rot) & 255;
        s += tanhf((float)kr[d] + q[d]) * vbuf[d];
      }
    }
    {
      float x = s;
#pragma unroll
      for (int off = 16; off > 0; off >>= 1) x = fmaxf(x, __shfl_xor(x, off, 32));
      if (lane == 0) red[wave] = x;
      __syncthreads();
      float mx = red[0];
#pragma unroll
      for (int w = 1; w < 16; ++w) mx = fmaxf(mx, red[w]);
      float e  = __expf(s - mx);
      float se = e;
#pragma unroll
      for (int off = 16; off > 0; off >>= 1) se += __shfl_xor(se, off, 32);
      if (lane == 0) red[16 + wave] = se;
      __syncthreads();
      float den = red[16];
#pragma unroll
      for (int w = 1; w < 16; ++w) den += red[16 + w];
      sc[tid] = e / den;                 // alpha
    }
    __syncthreads();
    // 8) context: ctx[d] = sum_te alpha[te] * memory[te][d]  (L2-resident fp32)
    {
      int d = tid & 255, half = tid >> 8;
      const float* mcol = mb + (size_t)half * 256 * DIM + d;
      float acc = 0.0f;
#pragma unroll 4
      for (int te = 0; te < 256; ++te)
        acc += sc[half * 256 + te] * mcol[(size_t)te * DIM];
      red[tid] = acc;
    }
    __syncthreads();
    if (tid < 256) ctx[tid] = red[tid] + red[tid + 256];
    __syncthreads();
    // 9) attention layer: att = [h1, ctx] @ Wa   (concat is free: h1ctx)
    gemv_row0<512, 256, 0, 16>(h1ctx, Wap, nullptr, att);
    __syncthreads();
    // 10) output: y = att @ Wo + bo  -> global
    gemv_row0<256, 400, 0, 16>(att, Wop, bo, out + ((size_t)b * TDEC + t) * NOUT);
    __syncthreads();
  }
}

// ---------------------------------------------------------------------------
extern "C" void kernel_launch(void* const* d_in, const int* in_sizes, int n_in,
                              void* d_out, int out_size, void* d_ws, size_t ws_size,
                              hipStream_t stream) {
  const float* dec    = (const float*)d_in[0];
  const float* memory = (const float*)d_in[1];
  const float* W1  = (const float*)d_in[2];
  const float* b1  = (const float*)d_in[3];
  const float* W2  = (const float*)d_in[4];
  const float* b2  = (const float*)d_in[5];
  const float* k0  = (const float*)d_in[6];
  const float* r0  = (const float*)d_in[7];
  const float* bi0 = (const float*)d_in[8];
  const float* br0 = (const float*)d_in[9];
  const float* k1  = (const float*)d_in[10];
  const float* r1  = (const float*)d_in[11];
  const float* bi1 = (const float*)d_in[12];
  const float* br1 = (const float*)d_in[13];
  const float* Wq  = (const float*)d_in[14];
  const float* Wm  = (const float*)d_in[15];
  const float* v   = (const float*)d_in[16];
  const float* Wa  = (const float*)d_in[17];
  const float* Wo  = (const float*)d_in[18];
  const float* bo  = (const float*)d_in[19];
  float* out = (float*)d_out;

  char* ws = (char*)d_ws;
  size_t off = 0;
  auto take = [&](size_t elems) { __bf16* p = (__bf16*)(ws + off); off += elems * 2; return p; };

  struct PackDesc { const float* src; __bf16* dst; int K, KT, N; };
  __bf16* W1p = take((size_t)13 * 16 * 512);  // K=400 pad 416
  __bf16* W2p = take((size_t)8 * 8 * 512);
  __bf16* k0p = take((size_t)12 * 48 * 512);
  __bf16* r0p = take((size_t)8 * 48 * 512);
  __bf16* k1p = take((size_t)8 * 48 * 512);
  __bf16* r1p = take((size_t)8 * 48 * 512);
  __bf16* Wqp = take((size_t)8 * 16 * 512);
  __bf16* Wmp = take((size_t)8 * 16 * 512);
  __bf16* Wap = take((size_t)16 * 16 * 512);
  __bf16* Wop = take((size_t)8 * 25 * 512);
  __bf16* P1  = take((size_t)NB * TDEC * 256);   // prenet hidden (bf16)
  __bf16* P   = take((size_t)NB * TDEC * PREN);  // prenet output (bf16)
  __bf16* Kb  = take((size_t)NB * TENC * DIM);   // keys (bf16)

  const PackDesc packs[10] = {
      {W1, W1p, 400, 13, 256}, {W2, W2p, 256, 8, 128},
      {k0, k0p, 384, 12, 768}, {r0, r0p, 256, 8, 768},
      {k1, k1p, 256, 8, 768},  {r1, r1p, 256, 8, 768},
      {Wq, Wqp, 256, 8, 256},  {Wm, Wmp, 256, 8, 256},
      {Wa, Wap, 512, 16, 256}, {Wo, Wop, 256, 8, 400},
  };
  for (int i = 0; i < 10; ++i) {
    int NT = packs[i].N / 16;
    int total = packs[i].KT * NT * 512;
    pack_kernel<<<(total + 255) / 256, 256, 0, stream>>>(
        packs[i].src, packs[i].dst, packs[i].K, packs[i].KT, packs[i].N, total);
  }

  prenet1_kernel<<<(NB * TDEC) / 16, 512, 0, stream>>>(dec, W1p, b1, P1);
  prenet2_kernel<<<(NB * TDEC) / 16, 512, 0, stream>>>(P1, W2p, b2, P);
  keys_kernel<<<(NB * TENC) / 16, 512, 0, stream>>>(memory, Wmp, Kb);

  // 256KB dynamic LDS: per-WG keys slab (MI455X: 320KB LDS per workgroup)
  decoder_kernel<<<NB, 512, (size_t)TENC * DIM * sizeof(__bf16), stream>>>(
      P, Kb, memory,
      k0p, bi0, r0p, br0,
      k1p, bi1, r1p, br1,
      Wqp, v, Wap, Wop, bo, out);
}
